// Policy_73916387164829
// MI455X (gfx1250) — matrix operand order repaired and checked by reading, not compiled, not run on hardware
//
#include <hip/hip_runtime.h>

// ---------------------------------------------------------------------------
// Policy net fused forward for gfx1250 (MI455X), fp16 WMMA + LDS-resident im2col
// ---------------------------------------------------------------------------

typedef __attribute__((ext_vector_type(16))) _Float16 v16h;
typedef __attribute__((ext_vector_type(8)))  _Float16 v8h;
typedef __attribute__((ext_vector_type(8)))  float    v8f;
typedef __attribute__((ext_vector_type(4)))  unsigned int u32x4;

#define BATCH      16384
#define MTOK       200
#define NLAY       22
#define K1         576            // 22*25=550 padded to 18*32
#define K2         1152           // 128*9 = 36*32
#define CH         128
#define HID        512

// LDS layout (bytes)
#define A1_OFF     0                         // [16][9][576] half  = 165888 B
#define A2_OFF     165888                    // [16][1152]   half  =  36864 B
#define A3_OFF     (165888+36864)            // [16][128]    half  =   4096 B
#define ASF_OFF    (165888+36864+4096)       // [16][32]     half  =   1024 B
#define AHID_OFF   (165888+36864+4096+1024)  // [16][512]    half  =  16384 B
#define SMEM_BYTES (165888+36864+4096+1024+16384)   // 224256

// workspace layout (bytes)
#define WS_W1T   0         // [128][576] half
#define WS_W2T   147456    // [128][1152] half
#define WS_FCT   442368    // [256][128] half
#define WS_SLT   507904    // [256][32] half
#define WS_HDT   524288    // [32][512] half
#define WS_HDB   557056    // [32] float

__device__ __constant__ float c_maxvec[NLAY] = {
    9.0f, 1.0f, 1.0f, 10.0f, 3.0f, 254.0f, 1.0f, 1.0f, 235.0f, 8.0f, 9.0f,
    250.0f, 29.0f, 1.0f, 1.0f, 8.0f, 1.0f, 1.0f, 6.0f, 3.0f, 1.0f, 2.0f};

// --- weight repack: B^T layouts [N][K] in fp16, K zero-padded -------------
__global__ void policy_prep(const float* __restrict__ c1w, const float* __restrict__ c2w,
                            const float* __restrict__ fcw, const float* __restrict__ sw,
                            const float* __restrict__ a0w, const float* __restrict__ a1w,
                            const float* __restrict__ vw,  const float* __restrict__ a0b,
                            const float* __restrict__ a1b, const float* __restrict__ vb,
                            _Float16* __restrict__ w1t, _Float16* __restrict__ w2t,
                            _Float16* __restrict__ fct, _Float16* __restrict__ slt,
                            _Float16* __restrict__ hdt, float* __restrict__ hdb) {
    int i = blockIdx.x * blockDim.x + threadIdx.x;
    if (i < CH * K1) {                                  // conv1: [o][c*25+kh*5+kw] pad->576
        int o = i / K1, k = i % K1;
        w1t[i] = (k < 550) ? (_Float16)c1w[o * 550 + k] : (_Float16)0.0f;
        return;
    }
    i -= CH * K1;
    if (i < CH * K2) { w2t[i] = (_Float16)c2w[i]; return; }          // conv2 flat match
    i -= CH * K2;
    if (i < 256 * 128) { fct[i] = (_Float16)fcw[i]; return; }        // fc [256][128]
    i -= 256 * 128;
    if (i < 256 * 32) {                                              // self [256][22->32]
        int n = i / 32, k = i % 32;
        slt[i] = (k < NLAY) ? (_Float16)sw[n * NLAY + k] : (_Float16)0.0f;
        return;
    }
    i -= 256 * 32;
    if (i < 32 * HID) {                                              // heads [20->32][512]
        int r = i / HID, k = i % HID;
        float v = 0.0f;
        if (r < 9)        v = a0w[r * HID + k];
        else if (r < 19)  v = a1w[(r - 9) * HID + k];
        else if (r == 19) v = vw[k];
        hdt[i] = (_Float16)v;
        return;
    }
    i -= 32 * HID;
    if (i < 32) {
        float v = 0.0f;
        if (i < 9)        v = a0b[i];
        else if (i < 19)  v = a1b[i - 9];
        else if (i == 19) v = vb[0];
        hdb[i] = v;
    }
}

// --- WMMA fragment helpers (ISA §7.12.2 layouts) ---------------------------
__device__ __forceinline__ v16h ld_frag(const _Float16* p0, const _Float16* p1) {
    v8h lo = *(const v8h*)p0;
    v8h hi = *(const v8h*)p1;
    return __builtin_shufflevector(lo, hi, 0,1,2,3,4,5,6,7,8,9,10,11,12,13,14,15);
}

// A: 16x32 fp16 tile. lane holds row M=lane&15; halves {kb..kb+7, kb+16..kb+23}, kb=8*(lane>=16)
__device__ __forceinline__ v16h ld_a(const _Float16* a, int strideK, int kk, int lane) {
    int row = lane & 15;
    int kb  = kk * 32 + ((lane & 16) ? 8 : 0);
    const _Float16* p = a + row * strideK + kb;
    return ld_frag(p, p + 16);
}

// B^T stored [N][K]. lane holds col N=lane&15; halves {kb..kb+7, kb+16..kb+23}, kb=8*(lane>=16)
__device__ __forceinline__ v16h ld_b(const _Float16* bt, int strideK, int ntile, int kk, int lane) {
    int n  = ntile * 16 + (lane & 15);
    int kb = kk * 32 + ((lane & 16) ? 8 : 0);
    const _Float16* p = bt + n * strideK + kb;
    return ld_frag(p, p + 16);
}

__device__ __forceinline__ v8f gemm16(const _Float16* a, int strideA,
                                      const _Float16* bt, int strideB,
                                      int ntile, int ksteps, int lane) {
    v8f acc = {0.f, 0.f, 0.f, 0.f, 0.f, 0.f, 0.f, 0.f};
    for (int kk = 0; kk < ksteps; ++kk) {
        v16h av = ld_a(a, strideA, kk, lane);
        v16h bv = ld_b(bt, strideB, ntile, kk, lane);
        acc = __builtin_amdgcn_wmma_f32_16x16x32_f16(false, av, false, bv,
                                                     (short)0, acc, false, false);
    }
    return acc;
}

// --- fused forward: 1 block = 16 samples, 8 waves --------------------------
__global__ void __launch_bounds__(256)
policy_fused(const int* __restrict__ obs,
             const float* __restrict__ c1b, const float* __restrict__ c2b,
             const float* __restrict__ fcb, const float* __restrict__ sb,
             const _Float16* __restrict__ w1t, const _Float16* __restrict__ w2t,
             const _Float16* __restrict__ fct, const _Float16* __restrict__ slt,
             const _Float16* __restrict__ hdt, const float* __restrict__ hdb,
             float* __restrict__ out) {
    extern __shared__ char smem[];
    _Float16* A1   = (_Float16*)(smem + A1_OFF);    // [16][9][576] im2col for conv1
    _Float16* A2   = (_Float16*)(smem + A2_OFF);    // [16][1152]   im2col for conv2
    _Float16* A3   = (_Float16*)(smem + A3_OFF);    // [16][128]    conv2 out
    _Float16* ASF  = (_Float16*)(smem + ASF_OFF);   // [16][32]     self input
    _Float16* AHID = (_Float16*)(smem + AHID_OFF);  // [16][512]    hidden

    const int tid  = threadIdx.x;
    const int lane = tid & 31;
    const int w    = tid >> 5;           // 8 waves
    const int shi  = (lane & 16) ? 8 : 0;  // D-frag sample offset

    // ---- phase 0: zero all of LDS
    {
        u32x4 z = {0u, 0u, 0u, 0u};
        u32x4* sm = (u32x4*)smem;
        for (int i = tid; i < SMEM_BYTES / 16; i += 256) sm[i] = z;
    }
    __syncthreads();

    // ---- phase 1: token scatter directly into conv1 im2col patches
    for (int i = tid; i < 16 * MTOK; i += 256) {
        int bl = i / MTOK, t = i % MTOK;
        int gi = ((blockIdx.x * 16 + bl) * MTOK + t) * 3;
        int c = obs[gi], a = obs[gi + 1], v = obs[gi + 2];
        if (c == 255) c = 0;
        if (a == 255) a = 0;
        if (v == 255) v = 0;
        int x = (c >> 4) & 15, y = c & 15;
        if (a >= NLAY || x >= 11 || y >= 11) continue;
        _Float16 val = (_Float16)((float)v / c_maxvec[a]);
        for (int px = 0; px < 3; ++px) {
            int kx = x - 3 * px;
            if (kx < 0 || kx > 4) continue;
            for (int py = 0; py < 3; ++py) {
                int ky = y - 3 * py;
                if (ky < 0 || ky > 4) continue;
                A1[bl * (9 * K1) + (px * 3 + py) * K1 + a * 25 + kx * 5 + ky] = val;
            }
        }
    }
    __syncthreads();

    // ---- phase 2a: self input = center pixel (patch 4, K = c*25+12)
    for (int i = tid; i < 16 * 32; i += 256) {
        int s = i >> 5, cc = i & 31;
        ASF[i] = (cc < NLAY) ? A1[s * (9 * K1) + 4 * K1 + cc * 25 + 12] : (_Float16)0.0f;
    }
    // ---- phase 2b: conv1 — wave w owns out-channel tile w for all 9 patches
    {
        int cch = w * 16 + (lane & 15);
        float bias = c1b[cch];
        for (int p = 0; p < 9; ++p) {
            v8f acc = gemm16(A1 + p * K1, 9 * K1, w1t, K1, w, K1 / 32, lane);
            #pragma unroll
            for (int j = 0; j < 8; ++j) {
                int s = j + shi;
                A2[s * K2 + cch * 9 + p] = (_Float16)fmaxf(acc[j] + bias, 0.0f);
            }
        }
    }
    __syncthreads();

    // ---- phase 3: conv2 — [16,1152] x [1152,128]
    {
        int cch = w * 16 + (lane & 15);
        float bias = c2b[cch];
        v8f acc = gemm16(A2, K2, w2t, K2, w, K2 / 32, lane);
        #pragma unroll
        for (int j = 0; j < 8; ++j) {
            int s = j + shi;
            A3[s * CH + cch] = (_Float16)fmaxf(acc[j] + bias, 0.0f);
        }
    }
    __syncthreads();

    // ---- phase 4: fc (cnn_f -> hidden[256:512]) and self (-> hidden[0:256])
    for (int r = 0; r < 2; ++r) {
        int nt = w + 8 * r;
        int n  = nt * 16 + (lane & 15);
        {   // fc: [16,128] x [128,256]
            v8f acc = gemm16(A3, CH, fct, CH, nt, CH / 32, lane);
            float bias = fcb[n];
            #pragma unroll
            for (int j = 0; j < 8; ++j) {
                int s = j + shi;
                AHID[s * HID + 256 + n] = (_Float16)fmaxf(acc[j] + bias, 0.0f);
            }
        }
        {   // self: [16,32] x [32,256]
            v8f acc = gemm16(ASF, 32, slt, 32, nt, 1, lane);
            float bias = sb[n];
            #pragma unroll
            for (int j = 0; j < 8; ++j) {
                int s = j + shi;
                AHID[s * HID + n] = (_Float16)fmaxf(acc[j] + bias, 0.0f);
            }
        }
    }
    __syncthreads();

    // ---- phase 5: heads — [16,512] x [512, 32(pad)] -> scatter to d_out
    if (w < 2) {
        v8f acc = gemm16(AHID, HID, hdt, HID, w, HID / 32, lane);
        int col = w * 16 + (lane & 15);
        float bias = hdb[col];
        #pragma unroll
        for (int j = 0; j < 8; ++j) {
            int sg = blockIdx.x * 16 + j + shi;
            float val = acc[j] + bias;
            if (col < 9)       out[sg * 9 + col] = val;                      // logits0 [B,9]
            else if (col < 19) out[BATCH * 9 + sg * 10 + (col - 9)] = val;   // logits1 [B,10]
            else if (col == 19) out[BATCH * 19 + sg] = val;                  // value  [B,1]
        }
    }
}

extern "C" void kernel_launch(void* const* d_in, const int* in_sizes, int n_in,
                              void* d_out, int out_size, void* d_ws, size_t ws_size,
                              hipStream_t stream) {
    const int*   obs = (const int*)d_in[0];
    const float* c1w = (const float*)d_in[1];
    const float* c1b = (const float*)d_in[2];
    const float* c2w = (const float*)d_in[3];
    const float* c2b = (const float*)d_in[4];
    const float* fcw = (const float*)d_in[5];
    const float* fcb = (const float*)d_in[6];
    const float* sw  = (const float*)d_in[7];
    const float* sb  = (const float*)d_in[8];
    const float* a0w = (const float*)d_in[9];
    const float* a0b = (const float*)d_in[10];
    const float* a1w = (const float*)d_in[11];
    const float* a1b = (const float*)d_in[12];
    const float* vw  = (const float*)d_in[13];
    const float* vb  = (const float*)d_in[14];

    char* ws = (char*)d_ws;
    _Float16* w1t = (_Float16*)(ws + WS_W1T);
    _Float16* w2t = (_Float16*)(ws + WS_W2T);
    _Float16* fct = (_Float16*)(ws + WS_FCT);
    _Float16* slt = (_Float16*)(ws + WS_SLT);
    _Float16* hdt = (_Float16*)(ws + WS_HDT);
    float*    hdb = (float*)(ws + WS_HDB);

    // total repacked elements = 73728+147456+32768+8192+16384+32 = 278560
    int prepN = 278560;
    policy_prep<<<(prepN + 255) / 256, 256, 0, stream>>>(
        c1w, c2w, fcw, sw, a0w, a1w, vw, a0b, a1b, vb,
        w1t, w2t, fct, slt, hdt, hdb);

    policy_fused<<<BATCH / 16, 256, SMEM_BYTES, stream>>>(
        obs, c1b, c2b, fcb, sb, w1t, w2t, fct, slt, hdt, hdb, (float*)d_out);
}
